// LstmUpsampleModule_7017976562385
// MI455X (gfx1250) — matrix operand order repaired and checked by reading, not compiled, not run on hardware
//
#include <hip/hip_runtime.h>

// ---------------------------------------------------------------------------
// LSTM upsample on gfx1250 (MI455X): bf16 WMMA everywhere, fp32 accumulate.
//   H=256 hidden, S=128 independent sequences, N=512 time steps, UP=4.
//   Input projections precomputed per 64-step chunk (32MB, L2-resident);
//   recurrence done by a persistent per-16-seq workgroup with h in LDS
//   (ping-pong) and c in registers, one barrier per time step.
//   v3: waves_per_eu(4,4) on the scan so the allocator keeps the pipelined
//   registers live (occupancy is hard-capped at 4 waves/SIMD anyway), and
//   native v_tanh_f32-based activations on the critical path.
// ---------------------------------------------------------------------------

typedef __attribute__((ext_vector_type(16))) __bf16        v16bf;
typedef __attribute__((ext_vector_type(8)))  float         v8f;
typedef __attribute__((ext_vector_type(8)))  unsigned int  v8u;

#define HD      256      // hidden size
#define SQ      128      // independent sequences
#define NTIME   512      // time steps
#define UP      4
#define TC      64       // time chunk
#define KSTEPS  8        // H / 32
#define NTILES  64       // 4H / 16

// packed weight: [ntile][kstep][lane][8 dwords]
#define PW_ONE  ((size_t)NTILES * KSTEPS * 32 * 8 * 4)   // 524288 B per matrix

// workspace offsets (bytes)
#define OFF_PW    0x0UL          // 4 packed matrices, 2 MB
#define OFF_BIAS  0x200000UL     // 2 x 1024 fp32
#define OFF_HST   0x202000UL     // h state bf16 [2][128][256]
#define OFF_CST   0x222000UL     // c state fp32 [2][128][256]
#define OFF_CUR   0x262000UL     // current input bf16 [512][128][256]  (32 MB)
#define OFF_Y0C   0x2262000UL    // layer0 chunk output bf16 [64][128][256]
#define OFF_G     0x2662000UL    // gate buffer fp32 [64][128][1024]    (32 MB)

__device__ __forceinline__ unsigned short f2bf(float f) {
  unsigned int u = __builtin_bit_cast(unsigned int, f);
  unsigned int r = u + 0x7FFFu + ((u >> 16) & 1u);   // RNE
  return (unsigned short)(r >> 16);
}

__device__ __forceinline__ float fast_tanh(float x) {
#if __has_builtin(__builtin_amdgcn_tanhf)
  return __builtin_amdgcn_tanhf(x);
#elif __has_builtin(__builtin_amdgcn_tanh_f32)
  return __builtin_amdgcn_tanh_f32(x);
#else
  return tanhf(x);
#endif
}
__device__ __forceinline__ float fast_sig(float x) {
  return __builtin_fmaf(0.5f, fast_tanh(0.5f * x), 0.5f);
}

union QPair { uint4 q[2]; v8u v; };

// ---------------------------------------------------------------------------
// Pack W (fp32 [1024][256]) -> bf16 in the contiguous-k B-operand order used in this file:
//   for (ntile,kstep): lane holds col n = ntile*16+(lane&15),
//   VGPR v holds K = kstep*32 + (lane>>4)*16 + 2v, 2v+1.
// ---------------------------------------------------------------------------
__global__ void __launch_bounds__(256) k_pack(const float* __restrict__ W0,
                                              const float* __restrict__ W1,
                                              const float* __restrict__ W2,
                                              const float* __restrict__ W3,
                                              unsigned int* __restrict__ pw) {
  int tid   = blockIdx.x * 256 + threadIdx.x;       // 65536 threads
  int lane  = tid & 31;
  int grp   = tid >> 5;
  int kstep = grp & 7;
  int ntile = (grp >> 3) & 63;
  int wsel  = grp >> 9;                              // 0..3
  const float* W = (wsel == 0) ? W0 : (wsel == 1) ? W1 : (wsel == 2) ? W2 : W3;
  unsigned int* dst = pw + wsel * (PW_ONE / 4)
                         + ((size_t)((ntile * 8 + kstep) * 32 + lane)) * 8;
  int n     = ntile * 16 + (lane & 15);
  int kbase = kstep * 32 + (lane >> 4) * 16;
#pragma unroll
  for (int v = 0; v < 8; ++v) {
    int K = kbase + 2 * v;
    unsigned int lo = f2bf(W[(size_t)n * HD + K]);
    unsigned int hi = f2bf(W[(size_t)n * HD + K + 1]);
    dst[v] = lo | (hi << 16);
  }
}

__global__ void __launch_bounds__(256) k_bias(const float* __restrict__ bih0,
                                              const float* __restrict__ bhh0,
                                              const float* __restrict__ bih1,
                                              const float* __restrict__ bhh1,
                                              float* __restrict__ bias) {
  int i = blockIdx.x * 256 + threadIdx.x;  // 2048
  if (i < 1024) bias[i] = bih0[i] + bhh0[i];
  else          bias[i] = bih1[i - 1024] + bhh1[i - 1024];
}

// copy x -> out slot 0, x -> bf16 cur buffer, zero h/c state
__global__ void __launch_bounds__(256) k_init(const float* __restrict__ x,
                                              float* __restrict__ out,
                                              unsigned short* __restrict__ cur,
                                              unsigned short* __restrict__ hst,
                                              float* __restrict__ cst) {
  size_t i = (size_t)blockIdx.x * 256 + threadIdx.x;   // 16,777,216
  float v = x[i];
  int h = (int)(i & 255);
  int s = (int)((i >> 8) & 127);
  size_t n = i >> 15;
  out[(n * 512 + (size_t)s * 4) * 256 + h] = v;
  cur[i] = f2bf(v);
  if (i < 2 * SQ * HD) { hst[i] = 0; cst[i] = 0.f; }
}

// ---------------------------------------------------------------------------
// Big parallel GEMM: G[row][1024] = in_bf16[row][256] @ W^T + bias
//   rows = TC*SQ = 8192.  One wave per (mtile, 4 n-tiles). 1024 blocks x 256.
//   Compiler keeps this at ~54 VGPRs -> high occupancy hides load latency.
// ---------------------------------------------------------------------------
__global__ void __launch_bounds__(256) k_gemm(const unsigned short* __restrict__ in,
                                              const unsigned int* __restrict__ pw,
                                              const float* __restrict__ bias,
                                              float* __restrict__ G) {
  int lane  = threadIdx.x & 31;
  int w     = threadIdx.x >> 5;
  int wid   = blockIdx.x * 8 + w;      // 0..8191
  int mtile = wid >> 4;
  int ng    = wid & 15;
  int m     = lane & 15;
  int khalf = lane >> 4;
  int row   = mtile * 16 + m;

  // A fragments: contiguous-k order matching k_pack (lane=M, VGPR0-3: K=khalf*16+0..7,
  // VGPR4-7: K=khalf*16+8..15), per kstep.
  const unsigned short* arow = in + (size_t)row * HD;
  v8u a[KSTEPS];
#pragma unroll
  for (int ks = 0; ks < KSTEPS; ++ks) {
    QPair u;
    u.q[0] = *(const uint4*)(arow + ks * 32 + khalf * 16);
    u.q[1] = *(const uint4*)(arow + ks * 32 + khalf * 16 + 8);
    a[ks] = u.v;
  }

  const v8u* bp = (const v8u*)pw;
  v8f acc[4];
#pragma unroll
  for (int j = 0; j < 4; ++j) {
    float bv = bias[(ng * 4 + j) * 16 + m];
    acc[j] = (v8f){bv, bv, bv, bv, bv, bv, bv, bv};
  }

#pragma unroll
  for (int ks = 0; ks < KSTEPS; ++ks) {
#pragma unroll
    for (int j = 0; j < 4; ++j) {
      v8u b = bp[(size_t)((ng * 4 + j) * 8 + ks) * 32 + lane];
      acc[j] = __builtin_amdgcn_wmma_f32_16x16x32_bf16(
                   false, __builtin_bit_cast(v16bf, a[ks]),
                   false, __builtin_bit_cast(v16bf, b),
                   (short)0, acc[j], false, false);
    }
  }

#pragma unroll
  for (int j = 0; j < 4; ++j) {
    int col = (ng * 4 + j) * 16 + m;
    float* grow = G + (size_t)mtile * 16 * 1024 + col;
#pragma unroll
    for (int r = 0; r < 8; ++r) {
      grow[(size_t)(r + khalf * 8) * 1024] = acc[j][r];
    }
  }
}

// ---------------------------------------------------------------------------
// Sequential scan over a TC-step chunk of one LSTM layer.
//   grid = 8 (16-seq groups), block = 512 (16 waves).
//   Wave w owns gate tiles {g*16+w : g=0..3} -> hidden cols w*16..w*16+15,
//   so i,f,g,o for a hidden slot live in one wave: c in regs, no exchange.
//   h ping-pongs through LDS (bf16), one barrier per step.
//   G for step t+1 is prefetched during step t; B is double-buffered per
//   K-step.  waves_per_eu(4,4): occupancy is fixed at 4 waves/SIMD by the
//   grid shape, so budget registers for exactly that and keep the pipeline
//   registers live instead of re-loading.
// ---------------------------------------------------------------------------
__global__ void __launch_bounds__(512)
__attribute__((amdgpu_waves_per_eu(4, 4)))
k_scan(const float* __restrict__ G,
       const unsigned int* __restrict__ pwhh,
       unsigned short* __restrict__ hst,
       float* __restrict__ cst,
       unsigned short* __restrict__ ybf,
       float* __restrict__ yout,
       int t0, int pass) {
  __shared__ __align__(16) unsigned short hbuf[2][16][264];  // padded rows
  int lane  = threadIdx.x & 31;
  int w     = threadIdx.x >> 5;       // 0..15
  int sg    = blockIdx.x;             // 0..7
  int m     = lane & 15;
  int khalf = lane >> 4;
  int hcol  = w * 16 + m;             // hidden column owned by this lane

  // load h state (bf16) into ping buffer 0
  for (int i = threadIdx.x; i < 16 * HD; i += 512) {
    int rr = i >> 8, kk = i & 255;
    hbuf[0][rr][kk] = hst[(size_t)(sg * 16 + rr) * HD + kk];
  }
  // load c state into registers
  float creg[8];
#pragma unroll
  for (int r = 0; r < 8; ++r) {
    creg[r] = cst[(size_t)(sg * 16 + r + khalf * 8) * HD + hcol];
  }

  const v8u* bp = (const v8u*)pwhh;

  // preload G accumulator-init for t = 0
  v8f gnext[4];
  {
    const float* gbase = G + ((size_t)sg * 16) * 1024;
#pragma unroll
    for (int g = 0; g < 4; ++g) {
      int col = (g * 16 + w) * 16 + m;
#pragma unroll
      for (int r = 0; r < 8; ++r)
        gnext[g][r] = gbase[(size_t)(r + khalf * 8) * 1024 + col];
    }
  }
  __syncthreads();

  for (int t = 0; t < TC; ++t) {
    int rb = t & 1, wbuf = rb ^ 1;

    v8f acc[4];
#pragma unroll
    for (int g = 0; g < 4; ++g) acc[g] = gnext[g];

    // prefetch next step's G (independent of LDS / barrier)
    {
      int tn = (t + 1 < TC) ? t + 1 : t;
      const float* gbase = G + ((size_t)tn * SQ + sg * 16) * 1024;
#pragma unroll
      for (int g = 0; g < 4; ++g) {
        int col = (g * 16 + w) * 16 + m;
#pragma unroll
        for (int r = 0; r < 8; ++r)
          gnext[g][r] = gbase[(size_t)(r + khalf * 8) * 1024 + col];
      }
    }

    // A fragments from LDS (two ds_load_b128 per kstep)
    v8u a[KSTEPS];
#pragma unroll
    for (int ks = 0; ks < KSTEPS; ++ks) {
      QPair u;
      u.q[0] = *(const uint4*)&hbuf[rb][m][ks * 32 + khalf * 16];
      u.q[1] = *(const uint4*)&hbuf[rb][m][ks * 32 + khalf * 16 + 8];
      a[ks] = u.v;
    }

    // ks-major WMMA with double-buffered B
    v8u bcur[4], bnxt[4];
#pragma unroll
    for (int g = 0; g < 4; ++g)
      bcur[g] = bp[(size_t)((g * 16 + w) * 8 + 0) * 32 + lane];
#pragma unroll
    for (int ks = 0; ks < KSTEPS; ++ks) {
      if (ks + 1 < KSTEPS) {
#pragma unroll
        for (int g = 0; g < 4; ++g)
          bnxt[g] = bp[(size_t)((g * 16 + w) * 8 + ks + 1) * 32 + lane];
      }
#pragma unroll
      for (int g = 0; g < 4; ++g) {
        acc[g] = __builtin_amdgcn_wmma_f32_16x16x32_bf16(
                     false, __builtin_bit_cast(v16bf, a[ks]),
                     false, __builtin_bit_cast(v16bf, bcur[g]),
                     (short)0, acc[g], false, false);
      }
#pragma unroll
      for (int g = 0; g < 4; ++g) bcur[g] = bnxt[g];
    }

    // cell update + outputs (native v_tanh_f32 activations)
#pragma unroll
    for (int r = 0; r < 8; ++r) {
      int M = r + khalf * 8;
      float iv = fast_sig(acc[0][r]);
      float fv = fast_sig(acc[1][r]);
      float gv = fast_tanh(acc[2][r]);
      float ov = fast_sig(acc[3][r]);
      float cn = fv * creg[r] + iv * gv;
      creg[r] = cn;
      float hv = ov * fast_tanh(cn);
      unsigned short hb = f2bf(hv);
      hbuf[wbuf][M][hcol] = hb;
      int srow = sg * 16 + M;
      if (yout) {  // layer 1: fp32 to d_out slot `pass`, bf16 to cur buffer
        size_t n = (size_t)(t0 + t);
        yout[(n * 512 + (size_t)srow * 4 + pass) * 256 + hcol] = hv;
        ybf[(n * SQ + srow) * HD + hcol] = hb;
      } else {     // layer 0: bf16 chunk-local y0 buffer
        ybf[((size_t)t * SQ + srow) * HD + hcol] = hb;
      }
    }
    __syncthreads();
  }

  // write back state (TC even -> final h in buffer 0)
  for (int i = threadIdx.x; i < 16 * HD; i += 512) {
    int rr = i >> 8, kk = i & 255;
    hst[(size_t)(sg * 16 + rr) * HD + kk] = hbuf[TC & 1][rr][kk];
  }
#pragma unroll
  for (int r = 0; r < 8; ++r) {
    cst[(size_t)(sg * 16 + r + khalf * 8) * HD + hcol] = creg[r];
  }
}

// ---------------------------------------------------------------------------
extern "C" void kernel_launch(void* const* d_in, const int* in_sizes, int n_in,
                              void* d_out, int out_size, void* d_ws, size_t ws_size,
                              hipStream_t stream) {
  (void)in_sizes; (void)n_in; (void)out_size; (void)ws_size;
  const float* x    = (const float*)d_in[0];
  const float* Wih0 = (const float*)d_in[1];
  const float* Whh0 = (const float*)d_in[2];
  const float* bih0 = (const float*)d_in[3];
  const float* bhh0 = (const float*)d_in[4];
  const float* Wih1 = (const float*)d_in[5];
  const float* Whh1 = (const float*)d_in[6];
  const float* bih1 = (const float*)d_in[7];
  const float* bhh1 = (const float*)d_in[8];

  char* ws = (char*)d_ws;
  unsigned int*   pw   = (unsigned int*)(ws + OFF_PW);
  float*          bias = (float*)(ws + OFF_BIAS);
  unsigned short* hst  = (unsigned short*)(ws + OFF_HST);
  float*          cst  = (float*)(ws + OFF_CST);
  unsigned short* cur  = (unsigned short*)(ws + OFF_CUR);
  unsigned short* y0c  = (unsigned short*)(ws + OFF_Y0C);
  float*          G    = (float*)(ws + OFF_G);
  float*          out  = (float*)d_out;

  k_pack<<<256, 256, 0, stream>>>(Wih0, Whh0, Wih1, Whh1, pw);
  k_bias<<<8, 256, 0, stream>>>(bih0, bhh0, bih1, bhh1, bias);
  k_init<<<65536, 256, 0, stream>>>(x, out, cur, hst, cst);

  unsigned int* pwih0 = pw;
  unsigned int* pwhh0 = pw + PW_ONE / 4;
  unsigned int* pwih1 = pw + 2 * (PW_ONE / 4);
  unsigned int* pwhh1 = pw + 3 * (PW_ONE / 4);

  for (int pass = 1; pass < UP; ++pass) {
    for (int ch = 0; ch < NTIME / TC; ++ch) {
      int t0 = ch * TC;
      // layer 0: input projection + scan (state slot 0)
      k_gemm<<<1024, 256, 0, stream>>>(cur + (size_t)t0 * SQ * HD, pwih0, bias, G);
      k_scan<<<8, 512, 0, stream>>>(G, pwhh0, hst, cst, y0c, nullptr, t0, pass);
      // layer 1: projection of y0 + scan (state slot 1); writes out + cur
      k_gemm<<<1024, 256, 0, stream>>>(y0c, pwih1, bias + 1024, G);
      k_scan<<<8, 512, 0, stream>>>(G, pwhh1, hst + SQ * HD, cst + SQ * HD,
                                    cur, out, t0, pass);
    }
  }
}
